// Attention_2491081032413
// MI455X (gfx1250) — matrix operand order repaired and checked by reading, not compiled, not run on hardware
//
#include <hip/hip_runtime.h>

typedef __attribute__((ext_vector_type(16))) _Float16 v16h;
typedef __attribute__((ext_vector_type(8)))  _Float16 v8h;
typedef __attribute__((ext_vector_type(8)))  float    v8f;

#define B_ 2
#define T_ 2048
#define E_ 1024
#define H_ 16
#define S_ 64
#define BT_ (B_*T_)
#define INV4 0.17677669529663687f   /* 1024^(-0.25) */
#define LN_EPS 1e-5f

__device__ __forceinline__ v8f wmma16(v16h a, v16h b, v8f c) {
  // v_wmma_f32_16x16x32_f16: (neg_a, A, neg_b, B, c_mod, C, reuse_a, reuse_b)
  return __builtin_amdgcn_wmma_f32_16x16x32_f16(false, a, false, b, (short)0, c, false, false);
}

// A fragment: 16x32 (M x K), row-major source with stride lda (elements).
// Lane (m=l&15, h=l>>4) holds K = {8h..8h+7} then {16+8h..16+8h+7}: two
// contiguous 16-byte runs -> two b128 loads, no packing.
__device__ __forceinline__ v16h frag_a(const _Float16* base, int lda) {
  const int lane = threadIdx.x & 31;
  const int m = lane & 15, hf = lane >> 4;
  const _Float16* p = base + m*lda + 8*hf;
  const v8h lo = *(const v8h*)(p);
  const v8h hi = *(const v8h*)(p + 16);
  return __builtin_shufflevector(lo, hi, 0,1,2,3,4,5,6,7,8,9,10,11,12,13,14,15);
}

// B fragment from an N-major tile: B(k,n) = base[n*ld + k].
// Lane (n=l&15, h=l>>4) holds K = {8h..8h+7} then {16+8h..16+8h+7}: two 16-byte runs.
__device__ __forceinline__ v16h frag_bk(const _Float16* base, int ld) {
  const int lane = threadIdx.x & 31;
  const int n = lane & 15, hf = lane >> 4;
  const _Float16* p = base + n*ld + 8*hf;
  const v8h lo = *(const v8h*)(p);
  const v8h hi = *(const v8h*)(p + 16);
  return __builtin_shufflevector(lo, hi, 0,1,2,3,4,5,6,7,8,9,10,11,12,13,14,15);
}

// Per-lane async copy of 16 bytes global -> LDS (CDNA5, tracked by ASYNCcnt).
__device__ __forceinline__ void async_copy_b128(void* lds, const void* gptr) {
  const unsigned lds_off = (unsigned)(uintptr_t)lds;   // low 32 bits = LDS byte offset
  asm volatile("global_load_async_to_lds_b128 %0, %1, off"
               :: "v"(lds_off), "v"(gptr) : "memory");
}
__device__ __forceinline__ void wait_async0() {
  asm volatile("s_wait_asynccnt 0x0" ::: "memory");
}

// ---------------------------------------------------------------------------
// 128x64-tile GEMM, k-slab 64: C[128,64] = A[128,1024] @ W[:, n0:n0+64]
// 8 waves, each owns a 32x32 tile -> 8 WMMA per wave per barrier pair.
// mode 0: out fp32 = C + bias             (final projection -> d_out)
// mode 1: per-head LayerNorm over 64 cols, * E^-0.25, f16 out [B,H,T,S] (Q/K)
// mode 2: f16 out TRANSPOSED [B,H,S,T], no LN                          (V)
// ---------------------------------------------------------------------------
__global__ __launch_bounds__(256) void proj_gemm_kernel(
    const float* __restrict__ Af, const _Float16* __restrict__ Ah,
    const float* __restrict__ W, const float* __restrict__ bias,
    const float* __restrict__ lnw, const float* __restrict__ lnb,
    float* __restrict__ outF, _Float16* __restrict__ outH, int mode)
{
  __shared__ _Float16 sA [128][72];  // 128 rows x 64 k, padded (144B stride)
  __shared__ _Float16 sBt[64][72];   // W tile TRANSPOSED: 64 n x 64 k, padded
  __shared__ float    sC[128][65];
  __shared__ float    sMu[128], sRs[128];

  const int tid  = threadIdx.x;
  const int lane = tid & 31;
  const int w    = tid >> 5;          // 8 waves
  const int wm   = w & 3, wn = w >> 2;
  const int m0   = blockIdx.x * 128;
  const int n0   = blockIdx.y * 64;

  v8f acc[2][2] = {};

  const int arow = tid >> 1;              // 0..127
  const int acol = (tid & 1) * 32;        // this thread covers acol..acol+31
  const int brow = tid >> 2;              // W k-row 0..63
  const int bcol = (tid & 3) * 16;        // W n-cols bcol..bcol+15

  for (int kk = 0; kk < E_; kk += 64) {
    // ---- stage A tile 128x64 (convert fp32 -> f16 in flight), row-major ----
    if (Af) {
      const float* src = Af + (size_t)(m0 + arow)*E_ + kk + acol;
#pragma unroll
      for (int q = 0; q < 4; ++q) {
        const float4 f0 = *(const float4*)(src + q*8);
        const float4 f1 = *(const float4*)(src + q*8 + 4);
        v8h av;
        av[0]=(_Float16)f0.x; av[1]=(_Float16)f0.y; av[2]=(_Float16)f0.z; av[3]=(_Float16)f0.w;
        av[4]=(_Float16)f1.x; av[5]=(_Float16)f1.y; av[6]=(_Float16)f1.z; av[7]=(_Float16)f1.w;
        *(v8h*)&sA[arow][acol + q*8] = av;
      }
    } else {
      const _Float16* src = Ah + (size_t)(m0 + arow)*E_ + kk + acol;
#pragma unroll
      for (int q = 0; q < 4; ++q)
        *(v8h*)&sA[arow][acol + q*8] = *(const v8h*)(src + q*8);
    }

    // ---- stage W tile 64x64 transposed (n-major) so B frags are contiguous ----
    {
      const float* srcW = W + (size_t)(kk + brow)*E_ + n0 + bcol;
#pragma unroll
      for (int q = 0; q < 4; ++q) {
        const float4 f = *(const float4*)(srcW + q*4);
        sBt[bcol + q*4 + 0][brow] = (_Float16)f.x;
        sBt[bcol + q*4 + 1][brow] = (_Float16)f.y;
        sBt[bcol + q*4 + 2][brow] = (_Float16)f.z;
        sBt[bcol + q*4 + 3][brow] = (_Float16)f.w;
      }
    }
    // gfx1250 global_prefetch_b8 hints for the next k-slab
    if (kk + 64 < E_) {
      if (Af) __builtin_prefetch(Af + (size_t)(m0 + arow)*E_ + kk + 64 + acol, 0, 1);
      __builtin_prefetch(W + (size_t)(kk + 64 + brow)*E_ + n0 + bcol, 0, 1);
    }
    __syncthreads();

    // ---- 2 k-steps x (2 A-frags x 2 B-frags) = 8 WMMA per wave ----
#pragma unroll
    for (int ks = 0; ks < 2; ++ks) {
      const v16h a0 = frag_a(&sA[wm*32][ks*32], 72);
      const v16h a1 = frag_a(&sA[wm*32 + 16][ks*32], 72);
      const v16h b0 = frag_bk(&sBt[wn*32][ks*32], 72);
      const v16h b1 = frag_bk(&sBt[wn*32 + 16][ks*32], 72);
      acc[0][0] = wmma16(a0, b0, acc[0][0]);
      acc[0][1] = wmma16(a0, b1, acc[0][1]);
      acc[1][0] = wmma16(a1, b0, acc[1][0]);
      acc[1][1] = wmma16(a1, b1, acc[1][1]);
    }
    __syncthreads();
  }

  // spill accumulators to LDS (C layout: row = 8*half + r, col = lane&15)
  {
    const int n = lane & 15, hf = lane >> 4;
#pragma unroll
    for (int ai = 0; ai < 2; ++ai)
#pragma unroll
      for (int r = 0; r < 8; ++r) {
        sC[wm*32 + ai*16 + 8*hf + r][wn*32 + n]      = acc[ai][0][r];
        sC[wm*32 + ai*16 + 8*hf + r][wn*32 + 16 + n] = acc[ai][1][r];
      }
  }
  __syncthreads();

  if (mode == 1) {
    if (tid < 128) {
      float s = 0.f, s2 = 0.f;
#pragma unroll 8
      for (int c = 0; c < 64; ++c) { const float x = sC[tid][c]; s += x; s2 += x*x; }
      const float mu  = s * (1.f/64.f);
      const float var = s2 * (1.f/64.f) - mu*mu;
      sMu[tid] = mu;
      sRs[tid] = rsqrtf(var + LN_EPS);
    }
    __syncthreads();
  }

  const int row = tid >> 1;            // 0..127
  const int c0  = (tid & 1) * 32;      // 32 cols per thread
  const int gm  = m0 + row;
  if (mode == 0) {
#pragma unroll
    for (int j = 0; j < 32; ++j) {
      const int c = c0 + j;
      outF[(size_t)gm*E_ + n0 + c] = sC[row][c] + bias[n0 + c];
    }
  } else if (mode == 1) {
    const int b = gm / T_, t = gm % T_;
    const int h = blockIdx.y;                 // one N-tile == one head (S=64)
    _Float16* dst = outH + (((size_t)b*H_ + h)*T_ + t)*S_;
    const float mu = sMu[row], rs = sRs[row];
#pragma unroll
    for (int j = 0; j < 32; ++j) {
      const int c = c0 + j;
      const float x = (sC[row][c] - mu) * rs * lnw[c] + lnb[c];
      dst[c] = (_Float16)(x * INV4);
    }
  } else {
    // V: write transposed [B,H,S,T] so flash P@V B-frags are contiguous
    const int b = gm / T_, t = gm % T_;
    const int h = blockIdx.y;
    _Float16* dstT = outH + (((size_t)b*H_ + h)*S_)*T_ + t;
#pragma unroll
    for (int j = 0; j < 32; ++j) {
      const int c = c0 + j;
      dstT[(size_t)c*T_] = (_Float16)sC[row][c];
    }
  }
}

// ---------------------------------------------------------------------------
// Causal flash attention: one block = (b,h) x 64 query rows; 4 waves x 16 rows.
// K/V tiles staged with GLOBAL_LOAD_ASYNC_TO_LDS_B128 (ASYNCcnt), online
// softmax in fp32, all matmuls via v_wmma_f32_16x16x32_f16.
// ---------------------------------------------------------------------------
__global__ __launch_bounds__(128) void flash_attn_kernel(
    const _Float16* __restrict__ Qh, const _Float16* __restrict__ Kh,
    const _Float16* __restrict__ Vt,   // [B,H,S,T]
    _Float16* __restrict__ Oh)         // [B,T,E]
{
  __shared__ _Float16 sK [64][72];   // keys x s   (N-major for QK^T B-frags)
  __shared__ _Float16 sVt[64][72];   // s x keys   (N-major for P@V  B-frags)
  __shared__ _Float16 sP[4][16][72]; // per-wave P staging (wave-private)

  const int tid  = threadIdx.x;
  const int lane = tid & 31;
  const int w    = tid >> 5;           // 4 waves
  const int n    = lane & 15, hf = lane >> 4;
  const int bh   = blockIdx.y;
  const int q0   = blockIdx.x * 64;

  const _Float16* Qp = Qh + (size_t)bh * T_ * S_;
  const _Float16* Kp = Kh + (size_t)bh * T_ * S_;
  const _Float16* Vp = Vt + (size_t)bh * S_ * T_;

  // Q fragments for this wave's 16 rows, held in registers for the whole pass
  const v16h aQ0 = frag_a(Qp + (size_t)(q0 + w*16)*S_, S_);
  const v16h aQ1 = frag_a(Qp + (size_t)(q0 + w*16)*S_ + 32, S_);

  const v8f zeroF = {};
  float mrow[8], lrow[8];
  v8f accO[4];
#pragma unroll
  for (int r = 0; r < 8; ++r) { mrow[r] = -1e30f; lrow[r] = 0.f; }
#pragma unroll
  for (int j = 0; j < 4; ++j) accO[j] = zeroF;

  for (int k0 = 0; k0 <= q0; k0 += 64) {          // causal: only blocks <= q0
    // ---- async-stage K (row-major [key][s]) and V^T (row-major [s][key]) ----
#pragma unroll
    for (int p = 0; p < 4; ++p) {
      const int idx = tid + p*128;
      const int r = idx >> 3, c = (idx & 7)*8;
      async_copy_b128(&sK[r][c],  Kp + (size_t)(k0 + r)*S_ + c);
      async_copy_b128(&sVt[r][c], Vp + (size_t)r*T_ + k0 + c);
    }
    wait_async0();
    __syncthreads();

    // ---- S = Q @ K^T  (16 x 64 per wave, 8 WMMA) ----
    v8f sc[4];
#pragma unroll
    for (int nt = 0; nt < 4; ++nt) {
      sc[nt] = zeroF;
      sc[nt] = wmma16(aQ0, frag_bk(&sK[nt*16][0],  72), sc[nt]);
      sc[nt] = wmma16(aQ1, frag_bk(&sK[nt*16][32], 72), sc[nt]);
    }

    // causal mask only needed on the diagonal block
    if (k0 == q0) {
#pragma unroll
      for (int nt = 0; nt < 4; ++nt)
#pragma unroll
        for (int r = 0; r < 8; ++r)
          if (k0 + nt*16 + n > q0 + w*16 + 8*hf + r) sc[nt][r] = -1e30f;
    }

    // ---- online softmax: row max / rescale (rows live in 16-lane halves) ----
    float mnew[8], alpha[8];
#pragma unroll
    for (int r = 0; r < 8; ++r) {
      float t = fmaxf(fmaxf(sc[0][r], sc[1][r]), fmaxf(sc[2][r], sc[3][r]));
      t = fmaxf(t, __shfl_xor(t, 8, 32));
      t = fmaxf(t, __shfl_xor(t, 4, 32));
      t = fmaxf(t, __shfl_xor(t, 2, 32));
      t = fmaxf(t, __shfl_xor(t, 1, 32));
      mnew[r]  = fmaxf(mrow[r], t);
      alpha[r] = __expf(mrow[r] - mnew[r]);
      mrow[r]  = mnew[r];
    }

    // ---- P = exp(S - m); row sums; stage P (f16) into wave-private LDS ----
    _Float16* sPw = &sP[w][0][0];
#pragma unroll
    for (int r = 0; r < 8; ++r) {
      float rsum = 0.f;
#pragma unroll
      for (int nt = 0; nt < 4; ++nt) {
        const float p = __expf(sc[nt][r] - mnew[r]);
        rsum += p;
        sPw[(8*hf + r)*72 + nt*16 + n] = (_Float16)p;
      }
      rsum += __shfl_xor(rsum, 8, 32);
      rsum += __shfl_xor(rsum, 4, 32);
      rsum += __shfl_xor(rsum, 2, 32);
      rsum += __shfl_xor(rsum, 1, 32);
      lrow[r] = lrow[r]*alpha[r] + rsum;
    }

#pragma unroll
    for (int j = 0; j < 4; ++j)
#pragma unroll
      for (int r = 0; r < 8; ++r)
        accO[j][r] *= alpha[r];

    // ---- O += P @ V (8 WMMA); DS in-order within a wave, no barrier needed ----
    const v16h aP0 = frag_a(sPw, 72);
    const v16h aP1 = frag_a(sPw + 32, 72);
#pragma unroll
    for (int j = 0; j < 4; ++j) {
      accO[j] = wmma16(aP0, frag_bk(&sVt[j*16][0],  72), accO[j]);
      accO[j] = wmma16(aP1, frag_bk(&sVt[j*16][32], 72), accO[j]);
    }
    __syncthreads();
  }

  // ---- normalize and write f16 attention output as [B,T,E] ----
  const int b = bh / H_, h = bh % H_;
  const int qg = q0 + w*16 + 8*hf;
#pragma unroll
  for (int r = 0; r < 8; ++r) {
    const float inv = 1.f / lrow[r];
    _Float16* dst = Oh + ((size_t)b*T_ + qg + r)*E_ + h*S_;
#pragma unroll
    for (int j = 0; j < 4; ++j)
      dst[j*16 + n] = (_Float16)(accO[j][r] * inv);
  }
}

extern "C" void kernel_launch(void* const* d_in, const int* in_sizes, int n_in,
                              void* d_out, int out_size, void* d_ws, size_t ws_size,
                              hipStream_t stream) {
  (void)in_sizes; (void)n_in; (void)out_size; (void)ws_size;
  const float* queries = (const float*)d_in[0];
  const float* keys    = (const float*)d_in[1];
  const float* values  = (const float*)d_in[2];
  const float* Wq      = (const float*)d_in[3];
  const float* Wk      = (const float*)d_in[4];
  const float* Wv      = (const float*)d_in[5];
  const float* Wo      = (const float*)d_in[6];
  const float* bo      = (const float*)d_in[7];
  const float* qlnw    = (const float*)d_in[8];
  const float* qlnb    = (const float*)d_in[9];
  const float* klnw    = (const float*)d_in[10];
  const float* klnb    = (const float*)d_in[11];

  const size_t per = (size_t)B_ * H_ * T_ * S_;   // 4M f16 elems each
  _Float16* Qh  = (_Float16*)d_ws;
  _Float16* Kh  = Qh + per;
  _Float16* Vh  = Kh + per;                       // [B,H,S,T] (transposed)
  _Float16* Ohh = Vh + per;                       // attention out, [B,T,E] f16

  dim3 gp(BT_/128, E_/64);
  proj_gemm_kernel<<<gp, 256, 0, stream>>>(queries, nullptr, Wq, nullptr, qlnw, qlnb, nullptr, Qh, 1);
  proj_gemm_kernel<<<gp, 256, 0, stream>>>(keys,    nullptr, Wk, nullptr, klnw, klnb, nullptr, Kh, 1);
  proj_gemm_kernel<<<gp, 256, 0, stream>>>(values,  nullptr, Wv, nullptr, nullptr, nullptr, nullptr, Vh, 2);
  flash_attn_kernel<<<dim3(T_/64, B_*H_), 128, 0, stream>>>(Qh, Kh, Vh, Ohh);
  proj_gemm_kernel<<<gp, 256, 0, stream>>>(nullptr, Ohh, Wo, bo, nullptr, nullptr, (float*)d_out, nullptr, 0);
}
